// VointGraphAttention_19911468384716
// MI455X (gfx1250) — matrix operand order repaired and checked by reading, not compiled, not run on hardware
//
#include <hip/hip_runtime.h>
#include <hip/hip_bf16.h>

// Problem constants (from reference)
#define BSZ   4
#define CCH   64
#define PP    1024
#define VV    12
#define TPP   4                 // p positions per workgroup
#define COLS_E (TPP*VV)         // 48 embed columns per block
#define COLS_C (TPP*VV*VV)      // 576 graph-conv columns per block
#define NEGF  1e10f
#define EPSF  1e-5f

typedef __attribute__((ext_vector_type(16))) __bf16 v16bf;
typedef __attribute__((ext_vector_type(8)))  __bf16 v8bf;
typedef __attribute__((ext_vector_type(8)))  float  v8f;

// Build one 16x32 bf16 A-fragment (weights) from LDS, contiguous k order, same as B:
// lanes 0-15: M=0..15, elems 0-15 -> K=(lh*16)+0..15
__device__ __forceinline__ v16bf load_afrag(const __bf16* sW, int mt, int ks, int lane) {
    const int m  = lane & 15;
    const int lh = lane >> 4;
    const __bf16* base = sW + (mt*16 + m)*CCH + ks*32 + lh*16;
    v8bf lo = *(const v8bf*)(base);
    v8bf hi = *(const v8bf*)(base + 8);
    v16bf a;
#pragma unroll
    for (int e = 0; e < 8; ++e) { a[e] = lo[e]; a[e+8] = hi[e]; }
    return a;
}

__device__ __forceinline__ void load_weight_frags(const __bf16* sW, v16bf afr[4][2], int lane) {
#pragma unroll
    for (int mt = 0; mt < 4; ++mt)
#pragma unroll
        for (int ks = 0; ks < 2; ++ks)
            afr[mt][ks] = load_afrag(sW, mt, ks, lane);
}

// Embed: H[:,col] = mask * relu(scale * (W @ X[:,col]) + shift), stored bf16 [col][ch]
__device__ __forceinline__ void embed_stage(const __bf16* sX, __bf16* sH,
                                            const v16bf afr[4][2],
                                            const float* sScale, const float* sShift,
                                            const float* sMask, int lane, int wv)
{
    const int n  = lane & 15;
    const int lh = lane >> 4;
    for (int t = wv; t < (CCH/16)*(COLS_E/16); t += 8) {
        const int mt = t & 3, nt = t >> 2;
        const int col = nt*16 + n;
        v8f acc = {0.f,0.f,0.f,0.f,0.f,0.f,0.f,0.f};
#pragma unroll
        for (int ks = 0; ks < 2; ++ks) {
            v16bf bf = *(const v16bf*)(sX + col*CCH + ks*32 + lh*16);
            acc = __builtin_amdgcn_wmma_f32_16x16x32_bf16(false, afr[mt][ks],
                                                          false, bf,
                                                          (short)0, acc, false, false);
        }
        const float mfv = sMask[col];
        v8bf pk;
#pragma unroll
        for (int r = 0; r < 8; ++r) {
            const int o = mt*16 + lh*8 + r;
            float y = acc[r]*sScale[o] + sShift[o];
            y = y > 0.f ? y : 0.f;
            pk[r] = (__bf16)(y * mfv);
        }
        *(v8bf*)(sH + col*CCH + mt*16 + lh*8) = pk;
    }
}

// Graph conv: virtual X rows 0-31 = H[0:32, (p,v)]; rows 32-63 = H[32:64, (p,(v-w)%12)].
// nt-outer: one pair of B fragments feeds 8 WMMAs, issued as 4 independent K0 ops
// followed by 4 K1 ops so no two dependent WMMAs are adjacent (bf16 RAW hazard = 5).
// is_attn: relu output (>=0) max-reduced over v via signed-int ds atomic max into
//          sRed[pw][o] (init 0 is safe: mask[...,0]==1 and relu >= 0).
// else:    attention-weighted (sAw[pw][o]) sum over v via ds atomic add.
__device__ __forceinline__ void conv_stage(const __bf16* sH, const v16bf afr[4][2],
                                           const float* sScale, const float* sShift,
                                           const float* sMask, float* sRed,
                                           const float* sAw, bool is_attn,
                                           int lane, int wv)
{
    const int n  = lane & 15;
    const int lh = lane >> 4;
    for (int nt = wv; nt < COLS_C/16; nt += 8) {
        const int col = nt*16 + n;
        const int pl  = col / (VV*VV);
        const int rem = col % (VV*VV);
        const int v   = rem / VV;
        const int w   = rem % VV;
        int vn = v - w; vn += (vn < 0) ? VV : 0;
        const int cenc = pl*VV + v;
        const int nbrc = pl*VV + vn;
        v16bf b0 = *(const v16bf*)(sH + cenc*CCH + lh*16);        // K 0..31  (center ch 0..31)
        v16bf b1 = *(const v16bf*)(sH + nbrc*CCH + 32 + lh*16);   // K 32..63 (neighbor ch 32..63)
        v8f acc[4];
#pragma unroll
        for (int mt = 0; mt < 4; ++mt) {                          // 4 independent WMMAs
            v8f z = {0.f,0.f,0.f,0.f,0.f,0.f,0.f,0.f};
            acc[mt] = __builtin_amdgcn_wmma_f32_16x16x32_bf16(false, afr[mt][0], false, b0,
                                                              (short)0, z, false, false);
        }
#pragma unroll
        for (int mt = 0; mt < 4; ++mt)                            // deps are 4 ops apart
            acc[mt] = __builtin_amdgcn_wmma_f32_16x16x32_bf16(false, afr[mt][1], false, b1,
                                                              (short)0, acc[mt], false, false);
        const float mfv = sMask[cenc];        // mask over v (uniform across r, mt)
        const int   pw  = pl*VV + w;
        if (mfv > 0.5f) {                     // masked v contributes nothing either way
#pragma unroll
            for (int mt = 0; mt < 4; ++mt) {
                v8f awv;
                if (!is_attn)
                    awv = *(const v8f*)(sAw + pw*CCH + mt*16 + lh*8);
#pragma unroll
                for (int r = 0; r < 8; ++r) {
                    const int o = mt*16 + lh*8 + r;
                    float y = acc[mt][r]*sScale[o] + sShift[o];
                    y = y > 0.f ? y : 0.f;    // relu -> non-negative
                    if (is_attn)
                        atomicMax((int*)&sRed[pw*CCH + o], __float_as_int(y));
                    else
                        atomicAdd(&sRed[pw*CCH + o], awv[r] * y);
                }
            }
        }
    }
}

extern "C" __global__ __launch_bounds__(256)
void voint_graph_attention_kernel(
    const float* __restrict__ feats, const int* __restrict__ mask,
    const float* Wea, const float* bea, const float* gea, const float* beea,
    const float* rmea, const float* rvea,
    const float* Wem, const float* bem, const float* gem, const float* beem,
    const float* rmem, const float* rvem,
    const float* Wat, const float* gat, const float* beat,
    const float* rmat, const float* rvat,
    const float* Wmn, const float* gmn, const float* bemn,
    const float* rmmn, const float* rvmn,
    float* __restrict__ out)
{
    __shared__ __attribute__((aligned(32))) __bf16 sW [CCH*CCH];     // current weight, bf16
    __shared__ __attribute__((aligned(32))) __bf16 sX [COLS_E*CCH];  // feats tile  [col][ch]
    __shared__ __attribute__((aligned(32))) __bf16 sHa[COLS_E*CCH];  // embed A out [col][ch]
    __shared__ __attribute__((aligned(32))) __bf16 sHm[COLS_E*CCH];  // embed M out [col][ch]
    __shared__ __attribute__((aligned(32))) float sAmax[COLS_E*CCH]; // [pw][o]: max, then aw
    __shared__ __attribute__((aligned(32))) float sPool[COLS_E*CCH]; // [pw][o]: pooled acc
    __shared__ float sScEa[CCH], sShEa[CCH], sScEm[CCH], sShEm[CCH];
    __shared__ float sScAt[CCH], sShAt[CCH], sScMn[CCH], sShMn[CCH];
    __shared__ float sMask[COLS_E];
    __shared__ float sInvCard[TPP];

    const int tid  = threadIdx.x;
    const int lane = tid & 31;
    const int wv   = tid >> 5;
    const int b    = blockIdx.x / (PP/TPP);
    const int p0   = (blockIdx.x % (PP/TPP)) * TPP;

    // Fold BN (+bias for embeds) into per-channel scale/shift; load mask row + 1/card.
    if (tid < CCH) {
        const int o = tid; float s;
        s = gea[o]*rsqrtf(rvea[o]+EPSF); sScEa[o]=s; sShEa[o]=s*(bea[o]-rmea[o])+beea[o];
        s = gem[o]*rsqrtf(rvem[o]+EPSF); sScEm[o]=s; sShEm[o]=s*(bem[o]-rmem[o])+beem[o];
        s = gat[o]*rsqrtf(rvat[o]+EPSF); sScAt[o]=s; sShAt[o]=beat[o]-s*rmat[o];
        s = gmn[o]*rsqrtf(rvmn[o]+EPSF); sScMn[o]=s; sShMn[o]=bemn[o]-s*rmmn[o];
    } else if (tid < CCH + TPP) {
        const int pl = tid - CCH;
        float cnt = 0.f;
        for (int v = 0; v < VV; ++v) {
            const float m = (float)mask[(b*PP + p0 + pl)*VV + v];
            sMask[pl*VV + v] = m;
            cnt += m;
        }
        sInvCard[pl] = 1.f / cnt;   // mask[...,0]==1 so cnt >= 1
    }
    if (lane == 0)  // warm L2 for the residual re-read at the end
        __builtin_prefetch(feats + ((size_t)b*CCH*PP + (size_t)p0)*VV, 0, 1);

    // Stage weights (embed A) and feats tile into LDS as bf16.
    for (int f = tid; f < CCH*CCH; f += 256) sW[f] = (__bf16)Wea[f];
    for (int f = tid; f < CCH*COLS_E; f += 256) {
        const int i  = f / COLS_E, cv = f % COLS_E;
        const int pl = cv / VV,    v  = cv % VV;
        sX[cv*CCH + i] = (__bf16)feats[((b*CCH + i)*PP + p0 + pl)*VV + v];
    }
    __syncthreads();

    v16bf afr[4][2];

    // Embed A -> sHa
    load_weight_frags(sW, afr, lane);
    embed_stage(sX, sHa, afr, sScEa, sShEa, sMask, lane, wv);
    __syncthreads();

    // Embed M -> sHm ; init amax buffer (0 is a valid identity: relu >= 0, v=0 unmasked)
    for (int f = tid; f < CCH*CCH; f += 256) sW[f] = (__bf16)Wem[f];
    __syncthreads();
    load_weight_frags(sW, afr, lane);
    embed_stage(sX, sHm, afr, sScEm, sShEm, sMask, lane, wv);
    for (int f = tid; f < COLS_E*CCH; f += 256) sAmax[f] = 0.f;
    __syncthreads();

    // Attention graph conv -> masked max over v (LDS atomic max, [pw][o])
    for (int f = tid; f < CCH*CCH; f += 256) sW[f] = (__bf16)Wat[f];
    __syncthreads();
    load_weight_frags(sW, afr, lane);
    conv_stage(sHa, afr, sScAt, sShAt, sMask, sAmax, sAmax, true, lane, wv);
    __syncthreads();

    // Masked softmax over w per (o, pl): 256 rows, one per thread. Zero pooled.
    {
        const int o = tid >> 2, pl = tid & (TPP-1);
        float tv[VV];
        float mx = -NEGF;
#pragma unroll
        for (int w = 0; w < VV; ++w) {
            const float mfw = sMask[pl*VV + w];
            tv[w] = (mfw > 0.5f) ? sAmax[(pl*VV + w)*CCH + o] : -NEGF;
            mx = fmaxf(mx, tv[w]);
        }
        float ssum = 0.f;
#pragma unroll
        for (int w = 0; w < VV; ++w) { tv[w] = __expf(tv[w] - mx); ssum += tv[w]; }
        const float inv = 1.f / ssum;
#pragma unroll
        for (int w = 0; w < VV; ++w) {
            const float mfw = sMask[pl*VV + w];
            sAmax[(pl*VV + w)*CCH + o] = (mfw > 0.5f) ? tv[w]*inv : 0.f;  // becomes aw
            sPool[(pl*VV + w)*CCH + o] = 0.f;
        }
    }
    for (int f = tid; f < CCH*CCH; f += 256) sW[f] = (__bf16)Wmn[f];
    __syncthreads();

    // Main graph conv -> attention-weighted pooling over v (LDS atomic add, [pw][o])
    load_weight_frags(sW, afr, lane);
    conv_stage(sHm, afr, sScMn, sShMn, sMask, sPool, sAmax, false, lane, wv);
    __syncthreads();

    // out = feats + mask * pooled / card
    for (int f = tid; f < CCH*COLS_E; f += 256) {
        const int o  = f / COLS_E, cv = f % COLS_E;
        const int pl = cv / VV,    w  = cv % VV;
        const float pooled = sPool[cv*CCH + o] * sInvCard[pl] * sMask[cv];
        const int g = ((b*CCH + o)*PP + p0 + pl)*VV + w;
        out[g] = feats[g] + pooled;
    }
}

extern "C" void kernel_launch(void* const* d_in, const int* in_sizes, int n_in,
                              void* d_out, int out_size, void* d_ws, size_t ws_size,
                              hipStream_t stream) {
    const float* feats = (const float*)d_in[0];
    const int*   mask  = (const int*)  d_in[1];
    const float* Wea = (const float*)d_in[2];
    const float* bea = (const float*)d_in[3];
    const float* gea = (const float*)d_in[4];
    const float* beea= (const float*)d_in[5];
    const float* rmea= (const float*)d_in[6];
    const float* rvea= (const float*)d_in[7];
    const float* Wem = (const float*)d_in[8];
    const float* bem = (const float*)d_in[9];
    const float* gem = (const float*)d_in[10];
    const float* beem= (const float*)d_in[11];
    const float* rmem= (const float*)d_in[12];
    const float* rvem= (const float*)d_in[13];
    const float* Wat = (const float*)d_in[14];
    const float* gat = (const float*)d_in[15];
    const float* beat= (const float*)d_in[16];
    const float* rmat= (const float*)d_in[17];
    const float* rvat= (const float*)d_in[18];
    const float* Wmn = (const float*)d_in[19];
    const float* gmn = (const float*)d_in[20];
    const float* bemn= (const float*)d_in[21];
    const float* rmmn= (const float*)d_in[22];
    const float* rvmn= (const float*)d_in[23];
    float* out = (float*)d_out;

    dim3 grid(BSZ * (PP / TPP));   // 1024 workgroups
    dim3 block(256);               // 8 wave32
    voint_graph_attention_kernel<<<grid, block, 0, stream>>>(
        feats, mask,
        Wea, bea, gea, beea, rmea, rvea,
        Wem, bem, gem, beem, rmem, rvem,
        Wat, gat, beat, rmat, rvat,
        Wmn, gmn, bemn, rmmn, rvmn,
        out);
}
